// DiscreteARTrajectoryHead_34385508171843
// MI455X (gfx1250) — matrix operand order repaired and checked by reading, not compiled, not run on hardware
//
#include <hip/hip_runtime.h>
#include <hip/hip_bf16.h>
#include <math.h>

// ---------------------------------------------------------------------------
// DiscreteARTrajectoryHead for MI455X (gfx1250, wave32, WMMA).
// All GEMMs run through v_wmma_f32_16x16x32_bf16 (bf16 A/B, f32 accum).
// GEMM: W tile staged via LDS (shared by all 8 waves), software-pipelined
// with register prefetch of the next k-step; each wave computes a 32x64 tile.
// All 8 B-fragment ds_loads are issued before the 8 WMMAs so a single
// dscnt wait covers them.  Weights/activations converted fp32->bf16 in
// workspace each launch (deterministic, graph-capture safe).
// ---------------------------------------------------------------------------

#define BB 64
#define NN 32
#define DD 512
#define TT 8
#define MM 20
#define VV 512
#define KK 8
#define HBB 16
#define FFF 2048
#define LL 2
#define HH 8
#define PP (HBB * HBB)
#define ROWS_EGO (BB * MM * TT)   // 10240
#define ROWS_BEV (BB * PP)        // 16384
#define ROWS_AKV (BB * KK * TT)   // 4096
#define ATT_SCALE 0.125f          // 1/sqrt(64)

typedef __bf16 bf16_t;
typedef bf16_t v16bf __attribute__((ext_vector_type(16)));
typedef bf16_t v8bf  __attribute__((ext_vector_type(8)));
typedef float  v8f   __attribute__((ext_vector_type(8)));

#define LDSPITCH 40   // bf16 elems per LDS row (80B = 5*16B: aligned, conflict-free)

// ---------------------------------------------------------------------------
// bf16 WMMA GEMM:  C[M,N] = A[M,K] * W[N,K]^T + bias
// Block: 256 threads = 8 waves; block tile 256(M) x 64(N); wave tile 32x64.
// W tile (64x32 bf16) staged in LDS per k-step; A fragments direct from
// global; both prefetched one k-step ahead.  Fragment layouts per CDNA5
// ISA 7.12.2 (16-bit A 16x32, B 32x16).
// ---------------------------------------------------------------------------
__global__ __launch_bounds__(256) void gemm_bf16_wmma(
    const bf16_t* __restrict__ A, const bf16_t* __restrict__ W,
    const float* __restrict__ bias, float* __restrict__ Cf,
    bf16_t* __restrict__ Cb, int Mrows, int Ncols, int Kdim)
{
    __shared__ __align__(16) bf16_t Wt[64 * LDSPITCH];

    const int tid  = threadIdx.x;
    const int lane = tid & 31;
    const int wave = tid >> 5;
    const int row0 = blockIdx.x * 256 + wave * 32;
    const int col0 = blockIdx.y * 64;

    // A fragment rows (two 16-row groups per wave), clamped for reads
    int mA0 = row0 + (lane & 15);
    int mA1 = mA0 + 16;
    if (mA0 >= Mrows) mA0 = Mrows - 1;
    if (mA1 >= Mrows) mA1 = Mrows - 1;
    const int aoff = (lane >> 4) << 3;        // A lane-half K offset (0 / 8)
    const bf16_t* arow0 = A + (size_t)mA0 * Kdim + aoff;
    const bf16_t* arow1 = A + (size_t)mA1 * Kdim + aoff;

    // W tile staging: thread -> (n = tid>>2, kchunk = (tid&3)*8)
    const int wn = tid >> 2;
    const int wk = (tid & 3) << 3;
    const bf16_t* wsrc = W + (size_t)(col0 + wn) * Kdim + wk;
    bf16_t* wdst = &Wt[wn * LDSPITCH + wk];

    // B fragment LDS read: row (lane&15)+f*16, k-offset (lane>>4)*8
    const int brow  = lane & 15;
    const int bboff = (lane >> 4) << 3;

    v8f acc[2][4] = {};

    // prologue: prefetch k-step 0
    v8bf wreg = *(const v8bf*)(wsrc);
    v8bf a0lo = *(const v8bf*)(arow0);
    v8bf a0hi = *(const v8bf*)(arow0 + 16);
    v8bf a1lo = *(const v8bf*)(arow1);
    v8bf a1hi = *(const v8bf*)(arow1 + 16);

    for (int kk = 0; kk < Kdim; kk += 32) {
        *(v8bf*)wdst = wreg;                 // ds_store_b128
        __syncthreads();

        v16bf afrag0 = __builtin_shufflevector(a0lo, a0hi,
            0, 1, 2, 3, 4, 5, 6, 7, 8, 9, 10, 11, 12, 13, 14, 15);
        v16bf afrag1 = __builtin_shufflevector(a1lo, a1hi,
            0, 1, 2, 3, 4, 5, 6, 7, 8, 9, 10, 11, 12, 13, 14, 15);

        if (kk + 32 < Kdim) {                // prefetch next k-step
            wreg = *(const v8bf*)(wsrc + kk + 32);
            a0lo = *(const v8bf*)(arow0 + kk + 32);
            a0hi = *(const v8bf*)(arow0 + kk + 48);
            a1lo = *(const v8bf*)(arow1 + kk + 32);
            a1hi = *(const v8bf*)(arow1 + kk + 48);
        }

        // Issue all 8 ds_load_b128 first -> one dscnt wait covers them all.
        v16bf bfrag[4];
#pragma unroll
        for (int f = 0; f < 4; ++f) {
            const bf16_t* pb = &Wt[(brow + f * 16) * LDSPITCH + bboff];
            v8bf blo = *(const v8bf*)(pb);       // ds_load_b128
            v8bf bhi = *(const v8bf*)(pb + 16);  // ds_load_b128
            bfrag[f] = __builtin_shufflevector(blo, bhi,
                0, 1, 2, 3, 4, 5, 6, 7, 8, 9, 10, 11, 12, 13, 14, 15);
        }
#pragma unroll
        for (int f = 0; f < 4; ++f) {
            acc[0][f] = __builtin_amdgcn_wmma_f32_16x16x32_bf16(
                false, afrag0, false, bfrag[f], (short)0, acc[0][f], false, false);
            acc[1][f] = __builtin_amdgcn_wmma_f32_16x16x32_bf16(
                false, afrag1, false, bfrag[f], (short)0, acc[1][f], false, false);
        }
        __syncthreads();                     // protect Wt before next store
    }

    // C/D layout: lane n = lane&15 ; VGPR r -> m = r + (lane>>4)*8
#pragma unroll
    for (int g = 0; g < 2; ++g) {
        const int smBase = row0 + g * 16 + ((lane >> 4) << 3);
#pragma unroll
        for (int f = 0; f < 4; ++f) {
            const int n = col0 + (lane & 15) + f * 16;
            const float bv = bias ? bias[n] : 0.0f;
#pragma unroll
            for (int r = 0; r < 8; ++r) {
                const int m = smBase + r;
                if (m < Mrows) {
                    float val = acc[g][f][r] + bv;
                    if (Cf) Cf[(size_t)m * Ncols + n] = val;
                    if (Cb) Cb[(size_t)m * Ncols + n] = (bf16_t)val;
                }
            }
        }
    }
}

// ---------------------------------------------------------------------------
// LayerNorm over D=512 (optional residual-add, ReLU, dual fp32/bf16 output)
// ---------------------------------------------------------------------------
__global__ __launch_bounds__(256) void ln_kernel(
    const float* __restrict__ X, const float* __restrict__ resid,
    const float* __restrict__ gamma, const float* __restrict__ beta_,
    float* __restrict__ Yf, bf16_t* __restrict__ Yb, int relu)
{
    const int row = blockIdx.x;
    const int tid = threadIdx.x;
    const float* x = X + (size_t)row * DD;
    float v0 = x[tid], v1 = x[tid + 256];
    if (resid) {
        const float* r = resid + (size_t)row * DD;
        v0 += r[tid]; v1 += r[tid + 256];
    }
    __shared__ float red[256];
    red[tid] = v0 + v1; __syncthreads();
    for (int off = 128; off; off >>= 1) {
        if (tid < off) red[tid] += red[tid + off];
        __syncthreads();
    }
    const float mean = red[0] * (1.0f / DD);
    __syncthreads();
    const float d0 = v0 - mean, d1 = v1 - mean;
    red[tid] = d0 * d0 + d1 * d1; __syncthreads();
    for (int off = 128; off; off >>= 1) {
        if (tid < off) red[tid] += red[tid + off];
        __syncthreads();
    }
    const float inv = rsqrtf(red[0] * (1.0f / DD) + 1e-5f);
    float o0 = d0 * inv * gamma[tid] + beta_[tid];
    float o1 = d1 * inv * gamma[tid + 256] + beta_[tid + 256];
    if (relu) { o0 = fmaxf(o0, 0.0f); o1 = fmaxf(o1, 0.0f); }
    if (Yf) { float* y = Yf + (size_t)row * DD; y[tid] = o0; y[tid + 256] = o1; }
    if (Yb) { bf16_t* y = Yb + (size_t)row * DD; y[tid] = (bf16_t)o0; y[tid + 256] = (bf16_t)o1; }
}

// ---------------------------------------------------------------------------
// Elementwise helpers
// ---------------------------------------------------------------------------
__global__ __launch_bounds__(256) void f2bf_kernel(
    const float* __restrict__ s, bf16_t* __restrict__ d, size_t n)
{
    for (size_t i = (size_t)blockIdx.x * 256 + threadIdx.x; i < n;
         i += (size_t)gridDim.x * 256)
        d[i] = (bf16_t)s[i];
}

__global__ __launch_bounds__(256) void gelu_to_bf16(
    const float* __restrict__ x, bf16_t* __restrict__ y, size_t n)
{
    for (size_t i = (size_t)blockIdx.x * 256 + threadIdx.x; i < n;
         i += (size_t)gridDim.x * 256) {
        float v = x[i];
        y[i] = (bf16_t)(0.5f * v * (1.0f + erff(v * 0.70710678118654752f)));
    }
}

// bev_feature [B, D, P] -> tokens [B*P, D] (transposed) as bf16
__global__ __launch_bounds__(256) void bev_tokens(
    const float* __restrict__ bevf, bf16_t* __restrict__ dst)
{
    const size_t n = (size_t)BB * PP * DD;
    for (size_t i = (size_t)blockIdx.x * 256 + threadIdx.x; i < n;
         i += (size_t)gridDim.x * 256) {
        int d = (int)(i & (DD - 1));
        size_t r = i >> 9;
        int p = (int)(r % PP);
        int b = (int)(r / PP);
        dst[i] = (bf16_t)bevf[((size_t)b * DD + d) * PP + p];
    }
}

// ---------------------------------------------------------------------------
// Top-K agent selection (+ all-invalid guard)  -- one block per batch
// ---------------------------------------------------------------------------
__global__ __launch_bounds__(32) void topk_kernel(
    const float* __restrict__ states, const float* __restrict__ labels,
    int* __restrict__ idx_out, int* __restrict__ inv_out)
{
    const int b = blockIdx.x, lane = threadIdx.x;
    __shared__ float dist[NN];
    __shared__ int vald[NN];
    __shared__ int sel[KK];
    const float lab = labels[b * NN + lane];
    const int valid = lab > -2.9444389791664403f;   // sigmoid(x) > 0.05
    const float x = states[(size_t)(b * NN + lane) * 8 + 0];
    const float y = states[(size_t)(b * NN + lane) * 8 + 1];
    dist[lane] = valid ? sqrtf(x * x + y * y) : INFINITY;
    vald[lane] = valid;
    __syncthreads();
    if (lane == 0) {
        for (int k = 0; k < KK; ++k) {
            float bd = INFINITY; int bi = 0;
            for (int j = 0; j < NN; ++j)
                if (dist[j] < bd) { bd = dist[j]; bi = j; }
            sel[k] = bi; dist[bi] = INFINITY;
            idx_out[b * KK + k] = bi;
        }
        int allinv = 1;
        for (int k = 0; k < KK; ++k) if (vald[sel[k]]) allinv = 0;
        for (int k = 0; k < KK; ++k)
            inv_out[b * KK + k] = (!vald[sel[k]]) && !allinv;
    }
}

__global__ __launch_bounds__(256) void agent_gather(
    const float* __restrict__ agents, const int* __restrict__ idx,
    bf16_t* __restrict__ dst)
{
    const size_t n = (size_t)BB * KK * DD;
    for (size_t i = (size_t)blockIdx.x * 256 + threadIdx.x; i < n;
         i += (size_t)gridDim.x * 256) {
        int d = (int)(i & (DD - 1));
        size_t r = i >> 9;
        int k = (int)(r % KK);
        int b = (int)(r / KK);
        dst[i] = (bf16_t)agents[((size_t)b * NN + idx[b * KK + k]) * DD + d];
    }
}

// agent_kv[b,k,t,:] = agent_enc[b,k,:] + step_e[t,:] + role_e[1,:]  (bf16)
__global__ __launch_bounds__(256) void build_akv(
    const float* __restrict__ enc, const float* __restrict__ step_e,
    const float* __restrict__ role1, bf16_t* __restrict__ out)
{
    const size_t n = (size_t)ROWS_AKV * DD;
    for (size_t i = (size_t)blockIdx.x * 256 + threadIdx.x; i < n;
         i += (size_t)gridDim.x * 256) {
        int d = (int)(i & (DD - 1));
        size_t r = i >> 9;
        int t = (int)(r % TT);
        size_t bk = r / TT;
        out[i] = (bf16_t)(enc[bk * DD + d] + step_e[t * DD + d] + role1[d]);
    }
}

// ---------------------------------------------------------------------------
// Sequential codebook NN scan.  gt_pos is broadcast over M -> tokens are
// identical for all modes; compute per-batch only: tokens[B,T].
// ---------------------------------------------------------------------------
__global__ __launch_bounds__(256) void cb_scan(
    const float* __restrict__ gt, const float* __restrict__ cb,
    int* __restrict__ tokens)
{
    const int b = blockIdx.x, tid = threadIdx.x;
    __shared__ float accx, accy;
    __shared__ float bestd[256];
    __shared__ int besti[256];
    if (tid == 0) { accx = 0.0f; accy = 0.0f; }
    __syncthreads();
    for (int t = 0; t < TT; ++t) {
        const float px = gt[(size_t)(b * TT + t) * 3 + 0];
        const float py = gt[(size_t)(b * TT + t) * 3 + 1];
        const float ax = accx, ay = accy;
        float bd = INFINITY; int bi = 0;
        for (int v = tid; v < VV; v += 256) {
            float dx = ax + cb[v * 2 + 0] - px;
            float dy = ay + cb[v * 2 + 1] - py;
            float dsq = dx * dx + dy * dy;
            if (dsq < bd) { bd = dsq; bi = v; }
        }
        bestd[tid] = bd; besti[tid] = bi;
        __syncthreads();
        for (int off = 128; off; off >>= 1) {
            if (tid < off) {
                float od = bestd[tid + off]; int oi = besti[tid + off];
                if (od < bestd[tid] || (od == bestd[tid] && oi < besti[tid])) {
                    bestd[tid] = od; besti[tid] = oi;
                }
            }
            __syncthreads();
        }
        if (tid == 0) {
            int ii = besti[0];
            tokens[b * TT + t] = ii;
            accx += cb[ii * 2 + 0]; accy += cb[ii * 2 + 1];
        }
        __syncthreads();
    }
}

// Teacher-forced input embedding (fp32 master + bf16 mirror)
__global__ __launch_bounds__(256) void build_ego(
    const int* __restrict__ tokens, const float* __restrict__ tok_emb,
    const float* __restrict__ bos_e, const float* __restrict__ ego_base,
    const float* __restrict__ step_e, const float* __restrict__ role0,
    const float* __restrict__ mode_e, float* __restrict__ yf,
    bf16_t* __restrict__ yb)
{
    const size_t n = (size_t)ROWS_EGO * DD;
    for (size_t i = (size_t)blockIdx.x * 256 + threadIdx.x; i < n;
         i += (size_t)gridDim.x * 256) {
        int d = (int)(i & (DD - 1));
        size_t r = i >> 9;
        int t = (int)(r % TT);
        int m = (int)((r / TT) % MM);
        int b = (int)(r / ((size_t)TT * MM));
        float base;
        if (t == 0) base = bos_e[d] + ego_base[(size_t)b * DD + d];
        else        base = tok_emb[(size_t)tokens[b * TT + t - 1] * DD + d];
        float v = base + step_e[t * DD + d] + role0[d] + mode_e[m * DD + d];
        yf[i] = v; yb[i] = (bf16_t)v;
    }
}

// ---------------------------------------------------------------------------
// Temporal causal self-attention (T=8).  One block per (b,m,head).
// qkv rows: [(b*M+m)*T + t][1536] = [q|k|v]
// ---------------------------------------------------------------------------
__global__ __launch_bounds__(64) void temporal_attn(
    const float* __restrict__ qkv, float* __restrict__ out)
{
    const int h = blockIdx.x & 7;
    const int bm = blockIdx.x >> 3;
    const float* base = qkv + (size_t)bm * TT * (3 * DD);
    __shared__ float Qs[TT][64], Ks[TT][64], Vs[TT][64], Ps[TT][TT];
    const int tid = threadIdx.x;
    for (int i = tid; i < TT * 64; i += 64) {
        int t = i >> 6, e = i & 63;
        Qs[t][e] = base[t * (3 * DD) + h * 64 + e];
        Ks[t][e] = base[t * (3 * DD) + DD + h * 64 + e];
        Vs[t][e] = base[t * (3 * DD) + 2 * DD + h * 64 + e];
    }
    __syncthreads();
    const int q = tid >> 3, k = tid & 7;
    float s = -INFINITY;
    if (k <= q) {
        s = 0.0f;
        for (int e = 0; e < 64; ++e) s += Qs[q][e] * Ks[k][e];
        s *= ATT_SCALE;
    }
    Ps[q][k] = s;
    __syncthreads();
    float rmax = -INFINITY;
    for (int j = 0; j <= q; ++j) rmax = fmaxf(rmax, Ps[q][j]);
    float rsum = 0.0f;
    for (int j = 0; j <= q; ++j) rsum += __expf(Ps[q][j] - rmax);
    const float p = (k <= q) ? __expf(s - rmax) / rsum : 0.0f;
    __syncthreads();
    Ps[q][k] = p;
    __syncthreads();
    for (int qq = 0; qq < TT; ++qq) {
        float o = 0.0f;
        for (int j = 0; j < TT; ++j) o += Ps[qq][j] * Vs[j][tid];
        out[((size_t)bm * TT + qq) * DD + h * 64 + tid] = o;
    }
}

// ---------------------------------------------------------------------------
// Ego->agent cross-attention.  One block per ego row (b,m,t); K=8 agents.
// akv rows: [(b*K+a)*T + t][1024] = [k|v]
// ---------------------------------------------------------------------------
__global__ __launch_bounds__(256) void agent_attn(
    const float* __restrict__ q, const float* __restrict__ akv,
    const int* __restrict__ inv, float* __restrict__ out)
{
    const int row = blockIdx.x;
    const int t = row % TT;
    const int b = row / (MM * TT);
    __shared__ float S[HH][KK], Pp[HH][KK];
    const int tid = threadIdx.x;
    if (tid < 64) {
        int h = tid >> 3, a = tid & 7;
        const float* qq = q + (size_t)row * DD + h * 64;
        const float* kp = akv + (((size_t)(b * KK + a) * TT + t) * 1024) + h * 64;
        float s = 0.0f;
        for (int e = 0; e < 64; ++e) s += qq[e] * kp[e];
        s *= ATT_SCALE;
        if (inv[b * KK + a]) s = -INFINITY;
        S[h][a] = s;
    }
    __syncthreads();
    if (tid < 64) {
        int h = tid >> 3, a = tid & 7;
        float rmax = -INFINITY;
        for (int j = 0; j < KK; ++j) rmax = fmaxf(rmax, S[h][j]);
        float rsum = 0.0f;
        for (int j = 0; j < KK; ++j) rsum += __expf(S[h][j] - rmax);
        Pp[h][a] = __expf(S[h][a] - rmax) / rsum;
    }
    __syncthreads();
    for (int d = tid; d < DD; d += 256) {
        int h = d >> 6, e = d & 63;
        float o = 0.0f;
        for (int a = 0; a < KK; ++a)
            o += Pp[h][a] *
                 akv[(((size_t)(b * KK + a) * TT + t) * 1024) + DD + h * 64 + e];
        out[(size_t)row * DD + d] = o;
    }
}

// ---------------------------------------------------------------------------
// Ego->BEV cross-attention.  One block per (ego row, head); P=256 keys.
// bkv rows: [b*P + p][1024] = [k|v]
// ---------------------------------------------------------------------------
__global__ __launch_bounds__(256) void bev_attn(
    const float* __restrict__ q, const float* __restrict__ bkv,
    float* __restrict__ out)
{
    const int h = blockIdx.x & 7;
    const int row = blockIdx.x >> 3;
    const int b = row / (MM * TT);
    __shared__ float Qs[64], S[PP], R[PP], O[256];
    const int tid = threadIdx.x;
    if (tid < 64) Qs[tid] = q[(size_t)row * DD + h * 64 + tid];
    __syncthreads();
    const float* kp = bkv + ((size_t)(b * PP + tid)) * 1024 + h * 64;
    float s = 0.0f;
    for (int e = 0; e < 64; ++e) s += Qs[e] * kp[e];
    s *= ATT_SCALE;
    R[tid] = s;
    __syncthreads();
    for (int off = 128; off; off >>= 1) {
        if (tid < off) R[tid] = fmaxf(R[tid], R[tid + off]);
        __syncthreads();
    }
    const float mx = R[0];
    __syncthreads();
    const float ex = __expf(s - mx);
    S[tid] = ex; R[tid] = ex;
    __syncthreads();
    for (int off = 128; off; off >>= 1) {
        if (tid < off) R[tid] += R[tid + off];
        __syncthreads();
    }
    const float denom = R[0];
    const int e = tid & 63, chunk = tid >> 6;
    float o = 0.0f;
    for (int j = 0; j < 64; ++j) {
        int key = chunk * 64 + j;
        o += S[key] * bkv[((size_t)(b * PP + key)) * 1024 + DD + h * 64 + e];
    }
    O[tid] = o;
    __syncthreads();
    if (tid < 64)
        out[(size_t)row * DD + h * 64 + tid] =
            (O[tid] + O[tid + 64] + O[tid + 128] + O[tid + 192]) / denom;
}

// ---------------------------------------------------------------------------
// Host launch
// ---------------------------------------------------------------------------
extern "C" void kernel_launch(void* const* d_in, const int* in_sizes, int n_in,
                              void* d_out, int out_size, void* d_ws,
                              size_t ws_size, hipStream_t stream)
{
    (void)in_sizes; (void)n_in; (void)out_size; (void)ws_size;

    const float* ego_query    = (const float*)d_in[0];
    const float* agents_query = (const float*)d_in[1];
    const float* bev_feature  = (const float*)d_in[2];
    const float* agent_states = (const float*)d_in[3];
    const float* agent_labels = (const float*)d_in[4];
    const float* gt_traj      = (const float*)d_in[5];
    const float* codebook     = (const float*)d_in[6];
    const float* ego_ctx_w = (const float*)d_in[7];  const float* ego_ctx_b = (const float*)d_in[8];
    const float* ego_ctx_g = (const float*)d_in[9];  const float* ego_ctx_beta = (const float*)d_in[10];
    const float* bevproj_w = (const float*)d_in[11]; const float* bevproj_b = (const float*)d_in[12];
    const float* bevproj_g = (const float*)d_in[13]; const float* bevproj_beta = (const float*)d_in[14];
    const float* agent_w = (const float*)d_in[15];   const float* agent_b = (const float*)d_in[16];
    const float* agent_g = (const float*)d_in[17];   const float* agent_beta = (const float*)d_in[18];
    const float* tok_emb = (const float*)d_in[19];
    const float* step_e  = (const float*)d_in[20];
    const float* mode_e  = (const float*)d_in[21];
    const float* role_e  = (const float*)d_in[22];
    const float* bos_e   = (const float*)d_in[23];
    const float* t_qkv_w = (const float*)d_in[24]; const float* t_qkv_b = (const float*)d_in[25];
    const float* t_out_w = (const float*)d_in[26]; const float* t_out_b = (const float*)d_in[27];
    const float* t_g = (const float*)d_in[28];     const float* t_beta = (const float*)d_in[29];
    const float* e_qkv_w = (const float*)d_in[30]; const float* e_qkv_b = (const float*)d_in[31];
    const float* e_out_w = (const float*)d_in[32]; const float* e_out_b = (const float*)d_in[33];
    const float* e_g = (const float*)d_in[34];     const float* e_beta = (const float*)d_in[35];
    const float* v_qkv_w = (const float*)d_in[36]; const float* v_qkv_b = (const float*)d_in[37];
    const float* v_out_w = (const float*)d_in[38]; const float* v_out_b = (const float*)d_in[39];
    const float* v_g = (const float*)d_in[40];     const float* v_beta = (const float*)d_in[41];
    const float* ffn_w1 = (const float*)d_in[42];  const float* ffn_b1 = (const float*)d_in[43];
    const float* ffn_w2 = (const float*)d_in[44];  const float* ffn_b2 = (const float*)d_in[45];
    const float* ffn_g = (const float*)d_in[46];   const float* ffn_beta = (const float*)d_in[47];
    const float* head_w = (const float*)d_in[48];  const float* head_b = (const float*)d_in[49];

    // ---- workspace carve (256B aligned bumps) ----
    uintptr_t cur = (uintptr_t)d_ws;
    auto take = [&](size_t elems, size_t esz) -> void* {
        cur = (cur + 255) & ~(uintptr_t)255;
        void* p = (void*)cur;
        cur += elems * esz;
        return p;
    };
    const size_t W_DD = (size_t)DD * DD;         // 512*512
    const size_t W_QKV = (size_t)LL * 3 * DD * DD;
    const size_t W_OUT = (size_t)LL * DD * DD;
    const size_t W_FFN = (size_t)LL * FFF * DD;

    bf16_t* wb_bev  = (bf16_t*)take(W_DD, 2);
    bf16_t* wb_agent= (bf16_t*)take(W_DD, 2);
    bf16_t* wb_ego  = (bf16_t*)take(W_DD, 2);
    bf16_t* wb_tqkv = (bf16_t*)take(W_QKV, 2);
    bf16_t* wb_eqkv = (bf16_t*)take(W_QKV, 2);
    bf16_t* wb_vqkv = (bf16_t*)take(W_QKV, 2);
    bf16_t* wb_tout = (bf16_t*)take(W_OUT, 2);
    bf16_t* wb_eout = (bf16_t*)take(W_OUT, 2);
    bf16_t* wb_vout = (bf16_t*)take(W_OUT, 2);
    bf16_t* wb_ffn1 = (bf16_t*)take(W_FFN, 2);
    bf16_t* wb_ffn2 = (bf16_t*)take(W_FFN, 2);
    bf16_t* wb_head = (bf16_t*)take(W_DD, 2);

    bf16_t* xb_bevraw = (bf16_t*)take((size_t)ROWS_BEV * DD, 2);
    bf16_t* xb_bev    = (bf16_t*)take((size_t)ROWS_BEV * DD, 2);
    bf16_t* xb_agent  = (bf16_t*)take((size_t)BB * KK * DD, 2);
    bf16_t* xb_egoq   = (bf16_t*)take((size_t)BB * DD, 2);
    bf16_t* xb_akv    = (bf16_t*)take((size_t)ROWS_AKV * DD, 2);
    bf16_t* xb_ego    = (bf16_t*)take((size_t)ROWS_EGO * DD, 2);
    bf16_t* xb_attn   = (bf16_t*)take((size_t)ROWS_EGO * DD, 2);
    bf16_t* xb_hid    = (bf16_t*)take((size_t)ROWS_EGO * FFF, 2);

    float* f_gemm   = (float*)take((size_t)ROWS_EGO * FFF, 4); // biggest GEMM out
    float* f_attn   = (float*)take((size_t)ROWS_EGO * DD, 4);
    float* f_proj   = (float*)take((size_t)ROWS_EGO * DD, 4);
    float* f_ego    = (float*)take((size_t)ROWS_EGO * DD, 4);
    float* f_akv    = (float*)take((size_t)ROWS_AKV * 2 * DD, 4);
    float* f_bkv    = (float*)take((size_t)ROWS_BEV * 2 * DD, 4);
    float* f_agt    = (float*)take((size_t)BB * KK * DD, 4);
    float* f_agenc  = (float*)take((size_t)BB * KK * DD, 4);
    float* f_egotmp = (float*)take((size_t)BB * DD, 4);
    float* f_egobase= (float*)take((size_t)BB * DD, 4);

    int* i_idx = (int*)take((size_t)BB * KK, 4);
    int* i_inv = (int*)take((size_t)BB * KK, 4);
    int* i_tok = (int*)take((size_t)BB * TT, 4);

    auto elgrid = [](size_t n) { return dim3((unsigned)((n + 255) / 256)); };
    auto cvt = [&](const float* s, bf16_t* d, size_t n) {
        f2bf_kernel<<<elgrid(n), 256, 0, stream>>>(s, d, n);
    };
    auto gemm = [&](const bf16_t* A, const bf16_t* W, const float* bias,
                    float* Cf, bf16_t* Cb, int Mr, int Nc, int Kd) {
        dim3 g((Mr + 255) / 256, Nc / 64);
        gemm_bf16_wmma<<<g, 256, 0, stream>>>(A, W, bias, Cf, Cb, Mr, Nc, Kd);
    };
    auto ln = [&](const float* X, const float* resid, const float* gm,
                  const float* bt, float* Yf, bf16_t* Yb, int rows, int relu) {
        ln_kernel<<<rows, 256, 0, stream>>>(X, resid, gm, bt, Yf, Yb, relu);
    };

    // ---- weight conversion ----
    cvt(bevproj_w, wb_bev, W_DD);   cvt(agent_w, wb_agent, W_DD);
    cvt(ego_ctx_w, wb_ego, W_DD);   cvt(t_qkv_w, wb_tqkv, W_QKV);
    cvt(e_qkv_w, wb_eqkv, W_QKV);   cvt(v_qkv_w, wb_vqkv, W_QKV);
    cvt(t_out_w, wb_tout, W_OUT);   cvt(e_out_w, wb_eout, W_OUT);
    cvt(v_out_w, wb_vout, W_OUT);   cvt(ffn_w1, wb_ffn1, W_FFN);
    cvt(ffn_w2, wb_ffn2, W_FFN);    cvt(head_w, wb_head, W_DD);

    // ---- BEV tokens: transpose -> GEMM -> LN+ReLU (bf16 only) ----
    bev_tokens<<<elgrid((size_t)ROWS_BEV * DD), 256, 0, stream>>>(bev_feature, xb_bevraw);
    gemm(xb_bevraw, wb_bev, bevproj_b, f_gemm, nullptr, ROWS_BEV, DD, DD);
    ln(f_gemm, nullptr, bevproj_g, bevproj_beta, nullptr, xb_bev, ROWS_BEV, 1);

    // ---- top-K agents, encode, build agent_kv ----
    topk_kernel<<<BB, 32, 0, stream>>>(agent_states, agent_labels, i_idx, i_inv);
    agent_gather<<<elgrid((size_t)BB * KK * DD), 256, 0, stream>>>(agents_query, i_idx, xb_agent);
    gemm(xb_agent, wb_agent, agent_b, f_agt, nullptr, BB * KK, DD, DD);
    ln(f_agt, nullptr, agent_g, agent_beta, f_agenc, nullptr, BB * KK, 1);
    build_akv<<<elgrid((size_t)ROWS_AKV * DD), 256, 0, stream>>>(f_agenc, step_e, role_e + DD, xb_akv);

    // ---- ego base ----
    cvt(ego_query, xb_egoq, (size_t)BB * DD);
    gemm(xb_egoq, wb_ego, ego_ctx_b, f_egotmp, nullptr, BB, DD, DD);
    ln(f_egotmp, nullptr, ego_ctx_g, ego_ctx_beta, f_egobase, nullptr, BB, 1);

    // ---- tokens + teacher-forced embedding ----
    cb_scan<<<BB, 256, 0, stream>>>(gt_traj, codebook, i_tok);
    build_ego<<<elgrid((size_t)ROWS_EGO * DD), 256, 0, stream>>>(
        i_tok, tok_emb, bos_e, f_egobase, step_e, role_e, mode_e, f_ego, xb_ego);

    // ---- transformer layers ----
    for (int i = 0; i < LL; ++i) {
        const size_t wq = (size_t)i * 3 * DD * DD;
        const size_t wo = (size_t)i * DD * DD;
        // 1) temporal causal self-attention
        gemm(xb_ego, wb_tqkv + wq, t_qkv_b + i * 3 * DD, f_gemm, nullptr,
             ROWS_EGO, 3 * DD, DD);
        temporal_attn<<<BB * MM * HH, 64, 0, stream>>>(f_gemm, f_attn);
        cvt(f_attn, xb_attn, (size_t)ROWS_EGO * DD);
        gemm(xb_attn, wb_tout + wo, t_out_b + i * DD, f_proj, nullptr,
             ROWS_EGO, DD, DD);
        ln(f_proj, f_ego, t_g + i * DD, t_beta + i * DD, f_ego, xb_ego, ROWS_EGO, 0);
        // 2) ego->agent cross-attention
        gemm(xb_ego, wb_eqkv + wq, e_qkv_b + i * 3 * DD, f_gemm, nullptr,
             ROWS_EGO, DD, DD);                                  // Q only
        gemm(xb_akv, wb_eqkv + wq + (size_t)DD * DD, e_qkv_b + i * 3 * DD + DD,
             f_akv, nullptr, ROWS_AKV, 2 * DD, DD);              // K|V
        agent_attn<<<ROWS_EGO, 256, 0, stream>>>(f_gemm, f_akv, i_inv, f_attn);
        cvt(f_attn, xb_attn, (size_t)ROWS_EGO * DD);
        gemm(xb_attn, wb_eout + wo, e_out_b + i * DD, f_proj, nullptr,
             ROWS_EGO, DD, DD);
        ln(f_proj, f_ego, e_g + i * DD, e_beta + i * DD, f_ego, xb_ego, ROWS_EGO, 0);
        // 3) ego->BEV cross-attention
        gemm(xb_ego, wb_vqkv + wq, v_qkv_b + i * 3 * DD, f_gemm, nullptr,
             ROWS_EGO, DD, DD);                                  // Q only
        gemm(xb_bev, wb_vqkv + wq + (size_t)DD * DD, v_qkv_b + i * 3 * DD + DD,
             f_bkv, nullptr, ROWS_BEV, 2 * DD, DD);              // K|V
        bev_attn<<<ROWS_EGO * HH, 256, 0, stream>>>(f_gemm, f_bkv, f_attn);
        cvt(f_attn, xb_attn, (size_t)ROWS_EGO * DD);
        gemm(xb_attn, wb_vout + wo, v_out_b + i * DD, f_proj, nullptr,
             ROWS_EGO, DD, DD);
        ln(f_proj, f_ego, v_g + i * DD, v_beta + i * DD, f_ego, xb_ego, ROWS_EGO, 0);
        // 4) FFN
        gemm(xb_ego, wb_ffn1 + (size_t)i * FFF * DD, ffn_b1 + i * FFF, f_gemm,
             nullptr, ROWS_EGO, FFF, DD);
        gelu_to_bf16<<<elgrid((size_t)ROWS_EGO * FFF), 256, 0, stream>>>(
            f_gemm, xb_hid, (size_t)ROWS_EGO * FFF);
        gemm(xb_hid, wb_ffn2 + (size_t)i * DD * FFF, ffn_b2 + i * DD, f_proj,
             nullptr, ROWS_EGO, DD, FFF);
        ln(f_proj, f_ego, ffn_g + i * DD, ffn_beta + i * DD, f_ego, xb_ego,
           ROWS_EGO, 0);
    }

    // ---- logit head: [10240,512] x head_w[512,512]^T -> d_out ----
    gemm(xb_ego, wb_head, head_b, (float*)d_out, nullptr, ROWS_EGO, VV, DD);
}
